// GCNLayer_HSI_pixel_36206574305751
// MI455X (gfx1250) — matrix operand order repaired and checked by reading, not compiled, not run on hardware
//
#include <hip/hip_runtime.h>

typedef __attribute__((ext_vector_type(16))) _Float16 v16h;
typedef __attribute__((ext_vector_type(8)))  _Float16 v8h;
typedef __attribute__((ext_vector_type(8)))  float    v8f;

#define NROWS 8192
#define DIN   200
#define DOUT  128
#define KC    64          // K-chunk staged in LDS per iteration
#define BSCALE 256.0f     // power-of-two scale keeps f16 B away from subnormals

// ---------------------------------------------------------------------------
// Pass 1: A = max(sigmoid(A),0.1) -> Aout (tuple output #2), dinv = rsqrt(rowsum)
// One block per row; 8192 fp32 per row, float4 streaming, LDS tree reduce.
// ---------------------------------------------------------------------------
__global__ __launch_bounds__(256) void k_sigdeg(const float* __restrict__ A,
                                                float* __restrict__ Aout,
                                                float* __restrict__ dinv) {
  const int row = blockIdx.x;
  const float4* __restrict__ ar = (const float4*)(A + (size_t)row * NROWS);
  float4* __restrict__ orow = (float4*)(Aout + (size_t)row * NROWS);
  float sum = 0.0f;
#pragma unroll
  for (int it = 0; it < 8; ++it) {
    const int i = it * 256 + threadIdx.x;
    float4 v = ar[i];
    float4 s;
    s.x = fmaxf(__builtin_amdgcn_rcpf(1.0f + __expf(-v.x)), 0.1f);
    s.y = fmaxf(__builtin_amdgcn_rcpf(1.0f + __expf(-v.y)), 0.1f);
    s.z = fmaxf(__builtin_amdgcn_rcpf(1.0f + __expf(-v.z)), 0.1f);
    s.w = fmaxf(__builtin_amdgcn_rcpf(1.0f + __expf(-v.w)), 0.1f);
    orow[i] = s;
    sum += (s.x + s.y) + (s.z + s.w);
  }
  __shared__ float red[256];
  red[threadIdx.x] = sum;
  __syncthreads();
#pragma unroll
  for (int s = 128; s > 0; s >>= 1) {
    if (threadIdx.x < (unsigned)s) red[threadIdx.x] += red[threadIdx.x + s];
    __syncthreads();
  }
  if (threadIdx.x == 0) dinv[row] = rsqrtf(red[0]);
}

// ---------------------------------------------------------------------------
// Pass 2: BT[c][r] = f16( 256 * dinv[r] * (H[r,:]@W[:,c] + b[c]) )
// One block per row r (128 threads = one column each); H row staged in LDS,
// W (100 KB) stays L2-resident across all 8192 blocks.
// ---------------------------------------------------------------------------
__global__ __launch_bounds__(128) void k_linear(const float* __restrict__ H,
                                                const float* __restrict__ W,
                                                const float* __restrict__ bias,
                                                const float* __restrict__ dinv,
                                                _Float16* __restrict__ BT) {
  __shared__ float hrow[DIN];
  const int r = blockIdx.x;
  const int c = threadIdx.x;
  for (int k = c; k < DIN; k += 128) hrow[k] = H[(size_t)r * DIN + k];
  __syncthreads();
  float acc = bias[c];
#pragma unroll 4
  for (int k = 0; k < DIN; ++k) acc = fmaf(hrow[k], W[k * DOUT + c], acc);
  BT[(size_t)c * NROWS + r] = (_Float16)(acc * (dinv[r] * BSCALE));
}

// ---------------------------------------------------------------------------
// Pass 3: out = LeakyReLU( diag(dinv) * (Asig @ BT^T) / 256 )
// 512 workgroups x 8 waves; 16-row M-strip per WG, one 16x16 N-tile per wave.
// A: fp32 global -> VGPR cvt -> LDS (needs conversion).
// B: fp16 global -> LDS via GLOBAL_LOAD_ASYNC_TO_LDS_B128 (ASYNCcnt path),
//    no VGPR round-trip. INST_OFFSET applies to both LDS and global address.
// ---------------------------------------------------------------------------
__global__ __launch_bounds__(256) void k_gemm(const float* __restrict__ Asig,
                                              const _Float16* __restrict__ BT,
                                              const float* __restrict__ dinv,
                                              float* __restrict__ out) {
  __shared__ _Float16 As[16 * KC];    //  2 KB: [row 0..15][k 0..63]
  __shared__ _Float16 Bs[DOUT * KC];  // 16 KB: [col 0..127][k 0..63]
  __shared__ float dloc[16];

  const int t = threadIdx.x;
  const int mbase = blockIdx.x * 16;
  if (t < 16) dloc[t] = dinv[mbase + t];

  const int wave = t >> 5;
  const int lane = t & 31;

  // A staging map: thread -> (row, 4 consecutive k)
  const int ar = t >> 4;                 // 0..15
  const int ac = (t & 15) * 4;           // 0..60
  const float* agp = Asig + (size_t)(mbase + ar) * NROWS + ac;

  // B staging map: thread -> (col, 32 consecutive k) = 64 bytes
  const int bc = t >> 1;                 // 0..127
  const int bk = (t & 1) * 32;           // 0 or 32
  const _Float16* bgp = BT + (size_t)bc * NROWS + bk;
  // LDS byte offset of this thread's Bs slice (generic->lds: low 32 bits)
  const unsigned bofs = (unsigned)(size_t)&Bs[bc * KC + bk];

  // WMMA fragment maps (wave32, contiguous k: lane half h holds K=16h..16h+15 for A and B)
  const int arow = lane & 15;
  const int aoff = (lane < 16) ? 0 : 16;  // A: lanes 0-15 hold K=0..15, 16-31 hold K=16..31
  const int bcol = (wave << 4) + (lane & 15);
  const int boff = (lane < 16) ? 0 : 16;  // B: lanes 0-15 hold K=0..15, 16-31 hold K=16..31

  v8f acc = {};

  for (int kb = 0; kb < NROWS; kb += KC) {
    __syncthreads();  // all waves done reading LDS before it is overwritten

    // --- stage B chunk: 64 bytes/thread straight into LDS (async, no VGPRs)
    const _Float16* bsrc = bgp + kb;
    asm volatile(
        "global_load_async_to_lds_b128 %0, %1, off\n\t"
        "global_load_async_to_lds_b128 %0, %1, off offset:16\n\t"
        "global_load_async_to_lds_b128 %0, %1, off offset:32\n\t"
        "global_load_async_to_lds_b128 %0, %1, off offset:48"
        :: "v"(bofs), "v"(bsrc) : "memory");

    // --- stage A chunk: 16x64 fp32 -> fp16 (conversion forces VGPR path)
    float4 av = *(const float4*)(agp + kb);
    _Float16* ad = &As[ar * KC + ac];
    ad[0] = (_Float16)av.x; ad[1] = (_Float16)av.y;
    ad[2] = (_Float16)av.z; ad[3] = (_Float16)av.w;

    asm volatile("s_wait_asynccnt 0x0" ::: "memory");  // B slices resident in LDS
    __syncthreads();                                   // + dscnt for A slices

#pragma unroll
    for (int sub = 0; sub < 2; ++sub) {
      const int k0 = sub * 32;
      v8h lo = *(const v8h*)&As[arow * KC + k0 + aoff];        // K off..off+7
      v8h hi = *(const v8h*)&As[arow * KC + k0 + aoff + 8];    // K off+8..off+15
      v16h a;
#pragma unroll
      for (int i = 0; i < 8; ++i) { a[i] = lo[i]; a[i + 8] = hi[i]; }
      v16h b = *(const v16h*)&Bs[bcol * KC + k0 + boff];       // K boff..boff+15
      acc = __builtin_amdgcn_wmma_f32_16x16x32_f16(
          /*neg_a=*/false, a, /*neg_b=*/false, b,
          /*c_mod=*/(short)0, acc, /*reuse_a=*/false, /*reuse_b=*/false);
    }
  }

  // Epilogue: C/D layout = VGPR v -> M = v (lanes 0-15) or v+8 (lanes 16-31), N = lane%16
  const int n = lane & 15;
  const int moff = (lane < 16) ? 0 : 8;
#pragma unroll
  for (int v = 0; v < 8; ++v) {
    const int m = v + moff;
    float val = acc[v] * dloc[m] * (1.0f / BSCALE);
    val = (val >= 0.0f) ? val : 0.01f * val;  // LeakyReLU
    out[(size_t)(mbase + m) * DOUT + (wave << 4) + n] = val;
  }
}

// ---------------------------------------------------------------------------
extern "C" void kernel_launch(void* const* d_in, const int* in_sizes, int n_in,
                              void* d_out, int out_size, void* d_ws, size_t ws_size,
                              hipStream_t stream) {
  (void)in_sizes; (void)n_in; (void)out_size; (void)ws_size;
  const float* H = (const float*)d_in[0];
  const float* A = (const float*)d_in[1];
  const float* W = (const float*)d_in[2];
  const float* b = (const float*)d_in[3];

  float* out  = (float*)d_out;                          // tuple output #1: [8192,128]
  float* Aout = (float*)d_out + (size_t)NROWS * DOUT;   // tuple output #2: [8192,8192]

  float*    dinv = (float*)d_ws;                        // 32 KB
  _Float16* BT   = (_Float16*)((char*)d_ws + 32768);    //  2 MB: [128][8192] fp16

  k_sigdeg<<<NROWS, 256, 0, stream>>>(A, Aout, dinv);
  k_linear<<<NROWS, 128, 0, stream>>>(H, W, b, dinv, BT);
  k_gemm<<<NROWS / 16, 256, 0, stream>>>(Aout, BT, dinv, out);
}
